// BiAttention_25340307046746
// MI455X (gfx1250) — matrix operand order repaired and checked by reading, not compiled, not run on hardware
//
#include <hip/hip_runtime.h>
#include <hip/hip_bf16.h>

// ---------------------------------------------------------------------------
// BiDAF bi-attention for MI455X (gfx1250, wave32, WMMA bf16 16x16x32).
// B=16, Lc=2048, Lq=256, D=256.  Q2C re-associated: S1 @ (S2t @ c).
// k_compute_T uses 2B scatters; other GEMM staging is 16B chunks: async global->LDS loads for bf16
// sources, TDM tensor_load_to_lds for the S1 A-tile, frag loads = ds_load_b128.
// ---------------------------------------------------------------------------

#define BN 16
#define LC 2048
#define LQ 256
#define DD 256
#define NEGV (-1e30f)

typedef __attribute__((ext_vector_type(16))) __bf16 v16bf;
typedef __attribute__((ext_vector_type(8)))  __bf16 v8bf;
typedef __attribute__((ext_vector_type(8)))  float  v8f;
typedef int v4i_g __attribute__((vector_size(16)));                 // gcc-style, matches builtin
typedef unsigned int u32x4 __attribute__((ext_vector_type(4)));
typedef int          i32x8 __attribute__((ext_vector_type(8)));
typedef int          i32x4 __attribute__((ext_vector_type(4)));

#if defined(__gfx1250__) && __has_builtin(__builtin_amdgcn_global_load_async_to_lds_b128)
#define HAVE_ASYNC_LDS 1
#else
#define HAVE_ASYNC_LDS 0
#endif

#if defined(__gfx1250__) && __has_builtin(__builtin_amdgcn_tensor_load_to_lds)
#define HAVE_TDM 1
#else
#define HAVE_TDM 0
#endif

__device__ __forceinline__ __bf16 to_bf16(float x) { return (__bf16)x; }

__device__ __forceinline__ v8f wmma_bf16(v16bf a, v16bf b, v8f c) {
  return __builtin_amdgcn_wmma_f32_16x16x32_bf16(false, a, false, b, (short)0, c,
                                                 false, false);
}

// 16B global -> LDS copy (async when available)
__device__ __forceinline__ void copy16_g2l(const __bf16* g, __bf16* l) {
#if HAVE_ASYNC_LDS
  __builtin_amdgcn_global_load_async_to_lds_b128(
      (__attribute__((address_space(1))) v4i_g*)g,
      (__attribute__((address_space(3))) v4i_g*)l, 0, 0);
#else
  *(v8bf*)l = *(const v8bf*)g;
#endif
}

__device__ __forceinline__ void async_fence_all() {
#if HAVE_ASYNC_LDS
#if __has_builtin(__builtin_amdgcn_s_wait_asynccnt)
  __builtin_amdgcn_s_wait_asynccnt(0);
#else
  asm volatile("s_wait_asynccnt 0" ::: "memory");
#endif
#endif
}

// Fragment-layout addressing (ISA 7.12.2):
// A (16x32, MxK): lane<16 -> row M=lane (kb=0), lane>=16 -> row M=lane-16 (kb=8)
//   element e<8 : K = kb+e ; e>=8 : K = kb+8+e
// B (32x16, KxN): lane<16 -> col N=lane (kb=0), lane>=16 -> col N=lane-16 (kb=8), same K map as A
__device__ __forceinline__ int a_lane(int m16, int k) { return (m16 & 15) + ((k & 8) << 1); }
__device__ __forceinline__ int a_elem(int k) { return (k & 7) + ((k & 16) >> 1); }
__device__ __forceinline__ int b_lane(int n16, int k) { return (n16 & 15) + ((k & 8) << 1); }
__device__ __forceinline__ int b_elem(int k) { return (k & 7) + ((k & 16) >> 1); }

// ---------------------------------------------------------------------------
// Kernel P: q -> bf16 (row-major) and bf16-transposed [B,DD,LQ]
// ---------------------------------------------------------------------------
__global__ __launch_bounds__(256) void k_prep(const float* __restrict__ q,
                                              __bf16* __restrict__ q16,
                                              __bf16* __restrict__ qT) {
  size_t idx = (size_t)blockIdx.x * 256 + threadIdx.x;  // BN*LQ*DD
  int d = (int)(idx & 255);
  int j = (int)((idx >> 8) & 255);
  int b = (int)(idx >> 16);
  float v  = q[idx];
  q16[idx] = to_bf16(v);
  qT[((size_t)b * DD + d) * LQ + j] = to_bf16(v);
}

// ---------------------------------------------------------------------------
// Kernel 0: rank-1 terms  s0[b,i] = c[b,i,:].w_c   s1[b,j] = q[b,j,:].w_q
// ---------------------------------------------------------------------------
__global__ __launch_bounds__(128) void k_rowdots(const float* __restrict__ c,
                                                 const float* __restrict__ q,
                                                 const float* __restrict__ c_w,
                                                 const float* __restrict__ q_w,
                                                 float* __restrict__ s0,
                                                 float* __restrict__ s1) {
  int wid  = (blockIdx.x * 128 + threadIdx.x) >> 5;
  int lane = threadIdx.x & 31;
  const int nr0 = BN * LC;
  const int nr1 = BN * LQ;
  float sum = 0.f;
  if (wid < nr0) {
    const float* row = c + (size_t)wid * DD;
#pragma unroll 4
    for (int k = lane; k < DD; k += 32) sum += row[k] * c_w[k];
  } else if (wid < nr0 + nr1) {
    const float* row = q + (size_t)(wid - nr0) * DD;
#pragma unroll 4
    for (int k = lane; k < DD; k += 32) sum += row[k] * q_w[k];
  }
#pragma unroll
  for (int off = 16; off; off >>= 1) sum += __shfl_xor(sum, off, 32);
  if (lane == 0) {
    if (wid < nr0) s0[wid] = sum;
    else if (wid < nr0 + nr1) s1[wid - nr0] = sum;
  }
}

// ---------------------------------------------------------------------------
// Kernel 1: S = (c .* w_cq) @ q^T + s0 + s1 + bias  -> raw S f32
//           fused masked row-softmax -> S1 bf16
// ---------------------------------------------------------------------------
__global__ __launch_bounds__(128) void k_compute_S(
    const float* __restrict__ c, const __bf16* __restrict__ q16,
    const float* __restrict__ q_mask, const float* __restrict__ cq_w,
    const float* __restrict__ s0, const float* __restrict__ s1,
    const float* __restrict__ bias, float* __restrict__ Sout,
    __bf16* __restrict__ S1out) {
  __shared__ __align__(32) __bf16 Af[4][32][16];   // frag layout
  __shared__ __align__(32) __bf16 Bf[16][32][16];  // frag layout
  __shared__ float qm[LQ];
  __shared__ float s1row[LQ];

  const int b    = blockIdx.y;
  const int m0   = blockIdx.x * 64;
  const int tid  = threadIdx.x;
  const int lane = tid & 31;
  const int w    = tid >> 5;

  for (int j = tid; j < LQ; j += 128) {
    qm[j]    = q_mask[b * LQ + j];
    s1row[j] = s1[b * LQ + j];
  }

  const v8f vz = {0.f, 0.f, 0.f, 0.f, 0.f, 0.f, 0.f, 0.f};
  v8f acc[16];
#pragma unroll
  for (int nt = 0; nt < 16; ++nt) acc[nt] = vz;

  const float*  cb  = c + (size_t)b * LC * DD;
  const __bf16* qb  = q16 + (size_t)b * LQ * DD;

  for (int kk = 0; kk < DD; kk += 32) {
    __syncthreads();
    // A: 256 chunks of 8 src f32 * cq_w -> 8 bf16 (frag layout, 16B store)
    for (int cidx = tid; cidx < 256; cidx += 128) {
      int strip = cidx >> 6;
      int l     = (cidx >> 1) & 31;
      int h     = cidx & 1;
      int m     = strip * 16 + (l & 15);
      int kb    = (l & 16) ? 8 : 0;
      int ks    = kk + kb + h * 16;
      const float* src = cb + (size_t)(m0 + m) * DD + ks;
      v8bf t;
#pragma unroll
      for (int e = 0; e < 8; ++e) t[e] = to_bf16(src[e] * cq_w[ks + e]);
      *(v8bf*)&Af[strip][l][h * 8] = t;
    }
    // B: per-lane payload = q16[j][kk+kb .. +7], q16[j][kk+kb+16 .. +23] -> async copies
    for (int cidx = tid; cidx < 512; cidx += 128) {
      int nt = cidx >> 5, l = cidx & 31;
      int j  = nt * 16 + (l & 15);
      int ks = kk + ((l & 16) ? 8 : 0);
      const __bf16* gsrc = qb + (size_t)j * DD + ks;
      __bf16*       ldst = &Bf[nt][l][0];
      copy16_g2l(gsrc, ldst);
      copy16_g2l(gsrc + 16, ldst + 8);
    }
    if (kk + 32 < DD)
      __builtin_prefetch(cb + (size_t)m0 * DD + kk + 32, 0, 1);
    async_fence_all();
    __syncthreads();
    v16bf a = *(const v16bf*)&Af[w][lane][0];
#pragma unroll
    for (int nt = 0; nt < 16; ++nt) {
      v16bf bv = *(const v16bf*)&Bf[nt][lane][0];
      acc[nt]  = wmma_bf16(a, bv, acc[nt]);
    }
  }

  // epilogue: rank-1 + bias, raw S store, fused masked row-softmax
  const int   nlo     = lane & 15;
  const int   mofs    = (lane & 16) ? 8 : 0;
  const int   rowbase = m0 + w * 16 + mofs;
  const float bia     = bias[0];
  float       s0v[8];
#pragma unroll
  for (int r = 0; r < 8; ++r) s0v[r] = s0[(size_t)b * LC + rowbase + r];

  float* Sb = Sout + ((size_t)b * LC + rowbase) * LQ;
#pragma unroll
  for (int nt = 0; nt < 16; ++nt) {
    int   n   = nlo + nt * 16;
    float s1v = s1row[n];
    float qmv = qm[n];
#pragma unroll
    for (int r = 0; r < 8; ++r) {
      float v                = acc[nt][r] + s0v[r] + s1v + bia;
      Sb[(size_t)r * LQ + n] = v;  // raw S (f32)
      acc[nt][r]             = v * qmv + (1.f - qmv) * NEGV;
    }
  }
  float rmax[8], rrcp[8];
#pragma unroll
  for (int r = 0; r < 8; ++r) {
    float m = acc[0][r];
#pragma unroll
    for (int nt = 1; nt < 16; ++nt) m = fmaxf(m, acc[nt][r]);
#pragma unroll
    for (int off = 1; off < 16; off <<= 1) m = fmaxf(m, __shfl_xor(m, off, 32));
    rmax[r] = m;
  }
#pragma unroll
  for (int r = 0; r < 8; ++r) {
    float s = 0.f;
#pragma unroll
    for (int nt = 0; nt < 16; ++nt) s += __expf(acc[nt][r] - rmax[r]);
#pragma unroll
    for (int off = 1; off < 16; off <<= 1) s += __shfl_xor(s, off, 32);
    rrcp[r] = 1.f / s;
  }
  __bf16* S1b = S1out + ((size_t)b * LC + rowbase) * LQ;
#pragma unroll
  for (int nt = 0; nt < 16; ++nt) {
    int n = nlo + nt * 16;
#pragma unroll
    for (int r = 0; r < 8; ++r)
      S1b[(size_t)r * LQ + n] = to_bf16(__expf(acc[nt][r] - rmax[r]) * rrcp[r]);
  }
}

// ---------------------------------------------------------------------------
// Kernel 2a/2b: masked column stats over i (online softmax, chunked + merge)
// ---------------------------------------------------------------------------
__global__ __launch_bounds__(256) void k_colstats1(const float* __restrict__ S,
                                                   const float* __restrict__ c_mask,
                                                   float* __restrict__ pm,
                                                   float* __restrict__ ps) {
  const int b = blockIdx.y, chunk = blockIdx.x, j = threadIdx.x;
  float     m = -1e38f, s = 0.f;
  for (int ii = 0; ii < 256; ++ii) {
    int   i  = chunk * 256 + ii;
    float cm = c_mask[b * LC + i];
    float x  = S[((size_t)b * LC + i) * LQ + j] * cm + (1.f - cm) * NEGV;
    float nm = fmaxf(m, x);
    s        = s * __expf(m - nm) + __expf(x - nm);
    m        = nm;
  }
  pm[((size_t)b * 8 + chunk) * LQ + j] = m;
  ps[((size_t)b * 8 + chunk) * LQ + j] = s;
}

__global__ __launch_bounds__(256) void k_colstats2(const float* __restrict__ pm,
                                                   const float* __restrict__ ps,
                                                   float* __restrict__ colmax,
                                                   float* __restrict__ colrcp) {
  const int b = blockIdx.x, j = threadIdx.x;
  float     m = -1e38f;
#pragma unroll
  for (int ch = 0; ch < 8; ++ch) m = fmaxf(m, pm[((size_t)b * 8 + ch) * LQ + j]);
  float s = 0.f;
#pragma unroll
  for (int ch = 0; ch < 8; ++ch)
    s += ps[((size_t)b * 8 + ch) * LQ + j] *
         __expf(pm[((size_t)b * 8 + ch) * LQ + j] - m);
  colmax[b * LQ + j] = m;
  colrcp[b * LQ + j] = 1.f / s;
}

// ---------------------------------------------------------------------------
// Kernel 3: T[b,j,d] = sum_i softmax_col(S^T)[j,i] * c[b,i,d]
// Output written TRANSPOSED: Tt[b,d,j]  (so k_final B staging is contiguous)
// ---------------------------------------------------------------------------
__global__ __launch_bounds__(128) void k_compute_T(
    const float* __restrict__ c, const float* __restrict__ S,
    const float* __restrict__ c_mask, const float* __restrict__ colmax,
    const float* __restrict__ colrcp, __bf16* __restrict__ Tt) {
  __shared__ __align__(32) __bf16 Af[4][32][16];
  __shared__ __align__(32) __bf16 Bf[4][32][16];
  __shared__ float cmx[64], crc[64];

  const int b    = blockIdx.z;
  const int j0   = blockIdx.x * 64;
  const int n0   = blockIdx.y * 64;
  const int tid  = threadIdx.x;
  const int lane = tid & 31;
  const int w    = tid >> 5;

  if (tid < 64) {
    cmx[tid] = colmax[b * LQ + j0 + tid];
    crc[tid] = colrcp[b * LQ + j0 + tid];
  }
  const v8f vz = {0.f, 0.f, 0.f, 0.f, 0.f, 0.f, 0.f, 0.f};
  v8f       acc[4] = {vz, vz, vz, vz};

  const float* cb  = c + (size_t)b * LC * DD;
  const float* Sb  = S + (size_t)b * LC * LQ;
  const float* cmb = c_mask + b * LC;
  __syncthreads();

  for (int kk = 0; kk < LC; kk += 32) {
    __syncthreads();
    // A: exp-scaled S^T scatter into frag layout (global reads coalesce on j)
    for (int idx = tid; idx < 64 * 32; idx += 128) {
      int   mj = idx & 63, k = idx >> 6;
      float cm = cmb[kk + k];
      float x  = Sb[(size_t)(kk + k) * LQ + j0 + mj] * cm + (1.f - cm) * NEGV;
      Af[mj >> 4][a_lane(mj, k)][a_elem(k)] =
          to_bf16(__expf(x - cmx[mj]) * crc[mj]);
    }
    // B: c rows scatter into frag layout (global reads coalesce on n)
    for (int idx = tid; idx < 32 * 64; idx += 128) {
      int n = idx & 63, k = idx >> 6;
      Bf[n >> 4][b_lane(n, k)][b_elem(k)] =
          to_bf16(cb[(size_t)(kk + k) * DD + n0 + n]);
    }
    if (kk + 32 < LC)
      __builtin_prefetch(Sb + (size_t)(kk + 32) * LQ + j0, 0, 1);
    __syncthreads();
    v16bf a = *(const v16bf*)&Af[w][lane][0];
#pragma unroll
    for (int nt = 0; nt < 4; ++nt) {
      v16bf bv = *(const v16bf*)&Bf[nt][lane][0];
      acc[nt]  = wmma_bf16(a, bv, acc[nt]);
    }
  }

  const int nlo  = lane & 15;
  const int mofs = (lane & 16) ? 8 : 0;
  const int jrow = j0 + w * 16 + mofs;
  __bf16*   Ttb  = Tt + (size_t)b * DD * LQ;
#pragma unroll
  for (int nt = 0; nt < 4; ++nt) {
    int d = n0 + nlo + nt * 16;
#pragma unroll
    for (int r = 0; r < 8; ++r)
      Ttb[(size_t)d * LQ + jrow + r] = to_bf16(acc[nt][r]);  // transposed store
  }
}

// ---------------------------------------------------------------------------
// Kernel 4: C2Q = S1 @ q, Q2C = S1 @ T (shared A frag), concat output.
// A tile (S1 bf16, row-major 64x32) staged by the Tensor Data Mover;
// Bq/Bt tiles staged by async global->LDS 16B copies from qT / Tt.
// ---------------------------------------------------------------------------
__global__ __launch_bounds__(128) void k_final(const float* __restrict__ c,
                                               const __bf16* __restrict__ qT,
                                               const __bf16* __restrict__ S1,
                                               const __bf16* __restrict__ Tt,
                                               float* __restrict__ out) {
  __shared__ __align__(32) __bf16 Afr[64][32];      // row-major (TDM target)
  __shared__ __align__(32) __bf16 Bqf[16][32][16];  // frag layout
  __shared__ __align__(32) __bf16 Btf[16][32][16];  // frag layout

  const int b    = blockIdx.y;
  const int m0   = blockIdx.x * 64;
  const int tid  = threadIdx.x;
  const int lane = tid & 31;
  const int w    = tid >> 5;

  const v8f vz = {0.f, 0.f, 0.f, 0.f, 0.f, 0.f, 0.f, 0.f};
  v8f       acc1[16], acc2[16];
#pragma unroll
  for (int nt = 0; nt < 16; ++nt) { acc1[nt] = vz; acc2[nt] = vz; }

  const __bf16* S1b = S1 + ((size_t)b * LC + m0) * LQ;
  const __bf16* qTb = qT + (size_t)b * DD * LQ;
  const __bf16* Ttb = Tt + (size_t)b * DD * LQ;

  for (int kk = 0; kk < LQ; kk += 32) {
    __syncthreads();
#if HAVE_TDM
    if (tid < 32) {
      if (lane == 0) {
        // D# per CDNA5 ISA: 2D tile 32(k) x 64(rows) of 2B
        unsigned long long ga = (unsigned long long)(size_t)(S1b + kk);
        unsigned int lds     = (unsigned int)(size_t)&Afr[0][0];
        u32x4 g0;
        g0[0] = 1u;                                   // count=1, user descriptor
        g0[1] = lds;                                  // lds_addr
        g0[2] = (unsigned int)ga;                     // global_addr[31:0]
        g0[3] = ((unsigned int)(ga >> 32) & 0x01FFFFFFu) | (2u << 30);  // +type=2
        i32x8 g1;
        g1[0] = 1 << 16;                              // data_size = 2B
        g1[1] = (32 & 0xFFFF) << 16;                  // tensor_dim0 = 32
        g1[2] = (64 & 0xFFFF) << 16;                  // tensor_dim1 = 64
        g1[3] = 32 << 16;                             // tile_dim0 = 32
        g1[4] = 64;                                   // tile_dim1 = 64, tile_dim2 = 0
        g1[5] = LQ;                                   // tensor_dim0_stride = 256
        g1[6] = 0;
        g1[7] = 0;
        i32x4 g2 = {0, 0, 0, 0};
        i32x4 g3 = {0, 0, 0, 0};
        i32x8 g4 = {0, 0, 0, 0, 0, 0, 0, 0};          // 6-arg toolchain variant
        __builtin_amdgcn_tensor_load_to_lds(g0, g1, g2, g3, g4, 0);
      }
      __builtin_amdgcn_s_wait_tensorcnt(0);
    }
#else
    for (int cidx = tid; cidx < 256; cidx += 128) {
      int m = cidx >> 2, hq = cidx & 3;
      *(v8bf*)&Afr[m][hq * 8] =
          *(const v8bf*)(S1b + (size_t)m * LQ + kk + hq * 8);
    }
#endif
    // Bq / Bt: per-lane payload = qT/Tt row chunks (contiguous 32B) -> async
    for (int cidx = tid; cidx < 512; cidx += 128) {
      int nt = cidx >> 5, l = cidx & 31;
      int d  = nt * 16 + (l & 15);
      int ks = kk + ((l & 16) ? 16 : 0);
      const __bf16* gq = qTb + (size_t)d * LQ + ks;
      const __bf16* gt = Ttb + (size_t)d * LQ + ks;
      __bf16* lq = &Bqf[nt][l][0];
      __bf16* lt = &Btf[nt][l][0];
      copy16_g2l(gq, lq);
      copy16_g2l(gq + 8, lq + 8);
      copy16_g2l(gt, lt);
      copy16_g2l(gt + 8, lt + 8);
    }
    async_fence_all();
    __syncthreads();
    // A frag from row-major tile: two contiguous 16B chunks per lane
    const __bf16* ap = &Afr[w * 16 + (lane & 15)][(lane & 16) ? 16 : 0];
    v8bf lo = *(const v8bf*)ap;
    v8bf hi = *(const v8bf*)(ap + 8);
    v16bf a = __builtin_shufflevector(lo, hi, 0, 1, 2, 3, 4, 5, 6, 7, 8, 9, 10,
                                      11, 12, 13, 14, 15);
#pragma unroll
    for (int nt = 0; nt < 16; ++nt) {
      v16bf b1 = *(const v16bf*)&Bqf[nt][lane][0];
      acc1[nt] = wmma_bf16(a, b1, acc1[nt]);
      v16bf b2 = *(const v16bf*)&Btf[nt][lane][0];
      acc2[nt] = wmma_bf16(a, b2, acc2[nt]);
    }
  }

  // epilogue: out = concat(c, C2Q, c*C2Q, c*Q2C)
  const int    nlo  = lane & 15;
  const int    mofs = (lane & 16) ? 8 : 0;
  const int    row  = m0 + w * 16 + mofs;
  const float* crow = c + ((size_t)b * LC + row) * DD;
  float*       ob   = out + ((size_t)b * LC + row) * (4 * DD);
#pragma unroll
  for (int nt = 0; nt < 16; ++nt) {
    int n = nlo + nt * 16;
#pragma unroll
    for (int r = 0; r < 8; ++r) {
      float  cv  = acc1[nt][r];  // placeholder to keep regs hot (overwritten)
      cv         = crow[(size_t)r * DD + n];
      float  c2q = acc1[nt][r];
      float  q2c = acc2[nt][r];
      size_t rb  = (size_t)r * 4 * DD;
      ob[rb + n]          = cv;
      ob[rb + DD + n]     = c2q;
      ob[rb + 2 * DD + n] = cv * c2q;
      ob[rb + 3 * DD + n] = cv * q2c;
    }
  }
}

// ---------------------------------------------------------------------------
extern "C" void kernel_launch(void* const* d_in, const int* in_sizes, int n_in,
                              void* d_out, int out_size, void* d_ws,
                              size_t ws_size, hipStream_t stream) {
  (void)in_sizes; (void)n_in; (void)out_size; (void)ws_size;
  const float* c      = (const float*)d_in[0];
  const float* q      = (const float*)d_in[1];
  const float* c_mask = (const float*)d_in[2];
  const float* q_mask = (const float*)d_in[3];
  const float* cq_w   = (const float*)d_in[4];
  const float* c_w    = (const float*)d_in[5];
  const float* q_w    = (const float*)d_in[6];
  const float* bias   = (const float*)d_in[7];
  float*       out    = (float*)d_out;

  char*   ws  = (char*)d_ws;
  float*  S   = (float*)(ws);              // 33.55 MB raw S f32
  __bf16* S1v = (__bf16*)(ws + 33554432);  // 16.78 MB S1 bf16
  __bf16* Tt  = (__bf16*)(ws + 50331648);  //  2.10 MB T^T bf16
  __bf16* q16 = (__bf16*)(ws + 52428800);  //  2.10 MB q bf16
  __bf16* qT  = (__bf16*)(ws + 54525952);  //  2.10 MB q^T bf16
  float*  s0  = (float*)(ws + 56623104);
  float*  s1  = (float*)(ws + 56754176);
  float*  pm  = (float*)(ws + 56770560);
  float*  ps  = (float*)(ws + 56901632);
  float*  cmx = (float*)(ws + 57032704);
  float*  crc = (float*)(ws + 57049088);

  k_prep<<<dim3(BN * LQ * DD / 256), 256, 0, stream>>>(q, q16, qT);
  const int nwaves = BN * LC + BN * LQ;  // 36864 rows
  k_rowdots<<<dim3(nwaves / 4), 128, 0, stream>>>(c, q, c_w, q_w, s0, s1);
  k_compute_S<<<dim3(LC / 64, BN), 128, 0, stream>>>(c, q16, q_mask, cq_w, s0,
                                                     s1, bias, S, S1v);
  k_colstats1<<<dim3(LC / 256, BN), 256, 0, stream>>>(S, c_mask, pm, ps);
  k_colstats2<<<dim3(BN), 256, 0, stream>>>(pm, ps, cmx, crc);
  k_compute_T<<<dim3(LQ / 64, DD / 64, BN), 128, 0, stream>>>(c, S, c_mask, cmx,
                                                              crc, Tt);
  k_final<<<dim3(LC / 64, BN), 128, 0, stream>>>(c, qT, S1v, Tt, out);
}
